// NNConvNet_11235634446409
// MI455X (gfx1250) — matrix operand order repaired and checked by reading, not compiled, not run on hardware
//
#include <hip/hip_runtime.h>
#include <hip/hip_bf16.h>
#include <hip/amd_detail/amd_hip_unsafe_atomics.h>

typedef _Float16 v16h __attribute__((ext_vector_type(16)));
typedef _Float16 v8h  __attribute__((ext_vector_type(8)));
typedef float    v8f  __attribute__((ext_vector_type(8)));

#define WAVES_PER_BLOCK 8
#define EDGE_BLOCK_THREADS 256
// per-wave scratch: xsT 4096 | h1s 2048 | h2s 2048 | efs 640 | ids 384 -> 9216
#define WAVE_SCRATCH 9216
// LDS: W3T 64KB | W2T 2KB | b3 4KB | 8 waves * 9216B
#define SMEM_EDGE (65536 + 2048 + 4096 + WAVES_PER_BLOCK * WAVE_SCRATCH)

__device__ __forceinline__ void wave_lds_fence() {
  // DS ops from one wave complete in order; waiting DScnt==0 makes all prior
  // LDS writes visible to all lanes of this wave. "memory" stops reordering.
  asm volatile("s_wait_dscnt 0x0" ::: "memory");
}

__device__ __forceinline__ v16h load_a_frag(const _Float16* row, int kb) {
  // 16-bit A 16x32 layout: lanes 0-15 K=0..7 & 16..23, lanes 16-31 K=8..15 & 24..31
  union { v16h v; v8h h[2]; } u;
  u.h[0] = *(const v8h*)(row + kb);
  u.h[1] = *(const v8h*)(row + kb + 16);
  return u.v;
}

__device__ __forceinline__ v16h load_b_frag(const _Float16* p) {
  // B 32x16: lane = column N, K=kb..kb+7 & kb+16..kb+23 (kb 0 or 8)
  union { v16h v; v8h h[2]; } u;
  u.h[0] = *(const v8h*)(p);
  u.h[1] = *(const v8h*)(p + 16);
  return u.v;
}

__device__ __forceinline__ v8f bias_frag(float b) {
  // C/D layout: N in lanes, M in VGPRs -> a column bias is lane-constant,
  // identical across all 8 C registers.
  v8f c;
#pragma unroll
  for (int v = 0; v < 8; v++) c[v] = b;
  return c;
}

__device__ __forceinline__ float selu_f(float v) {
  const float scale = 1.0507009873554805f, alpha = 1.6732632423543772f;
  return scale * (v > 0.f ? v : alpha * expm1f(v));
}

// ---------------- small prep / elementwise kernels ----------------

__global__ void prep_weights(const float* __restrict__ W2, const float* __restrict__ W3,
                             _Float16* __restrict__ w2t, _Float16* __restrict__ w3t) {
  int i = blockIdx.x * blockDim.x + threadIdx.x;
  const int nW2 = 3 * 32 * 32;
  const int nW3 = 3 * 32 * 1024;
  if (i < nW2) {
    int l = i / 1024, r = i % 1024, c = r / 32, k = r % 32;
    w2t[l * 1024 + c * 32 + k] = (_Float16)W2[l * 1024 + k * 32 + c];
  } else if (i < nW2 + nW3) {
    int j = i - nW2;
    int l = j / 32768, r = j % 32768, c = r / 32, k = r % 32;
    w3t[(size_t)l * 32768 + c * 32 + k] = (_Float16)W3[(size_t)l * 32768 + k * 1024 + c];
  }
}

__global__ void init_x_kernel(const float* __restrict__ node_attrs,
                              const float* __restrict__ embed_W,
                              const float* __restrict__ embed_b,
                              float* __restrict__ x, int N_) {
  int i = blockIdx.x * blockDim.x + threadIdx.x;
  if (i >= N_ * 32) return;
  int n = i >> 5, h = i & 31;
  x[i] = fmaf(node_attrs[n], embed_W[h], embed_b[h]);
}

__global__ void ef_kernel(const float* __restrict__ pos, const float* __restrict__ shifts,
                          const float* __restrict__ eattr, const int* __restrict__ eidx,
                          float* __restrict__ ef, int E_) {
  int e = blockIdx.x * blockDim.x + threadIdx.x;
  if (e >= E_) return;
  int s = eidx[e], d = eidx[E_ + e];
  float vx = pos[d * 3 + 0] - pos[s * 3 + 0] + shifts[(size_t)e * 3 + 0];
  float vy = pos[d * 3 + 1] - pos[s * 3 + 1] + shifts[(size_t)e * 3 + 1];
  float vz = pos[d * 3 + 2] - pos[s * 3 + 2] + shifts[(size_t)e * 3 + 2];
  float len = sqrtf(vx * vx + vy * vy + vz * vz);
  size_t o = (size_t)e * 5;
  ef[o + 0] = vx; ef[o + 1] = vy; ef[o + 2] = vz; ef[o + 3] = len; ef[o + 4] = eattr[e];
}

__global__ void init_agg_kernel(float* __restrict__ agg, const float* __restrict__ conv_b,
                                int layer, int N_) {
  int i = blockIdx.x * blockDim.x + threadIdx.x;
  if (i >= N_ * 32) return;
  agg[i] = conv_b[layer * 32 + (i & 31)];
}

__global__ void update_x_kernel(float* __restrict__ x, const float* __restrict__ agg, int N_) {
  int i = blockIdx.x * blockDim.x + threadIdx.x;
  if (i >= N_ * 32) return;
  x[i] += fmaxf(agg[i], 0.f);
}

__global__ void init_pool_kernel(float* __restrict__ pooled, float* __restrict__ counts) {
  int i = blockIdx.x * blockDim.x + threadIdx.x;
  if (i < 64 * 32) pooled[i] = 0.f;
  if (i < 64) counts[i] = 0.f;
}

__global__ void pool_kernel(const float* __restrict__ x, const int* __restrict__ batch,
                            float* __restrict__ pooled, float* __restrict__ counts, int N_) {
  int i = blockIdx.x * blockDim.x + threadIdx.x;
  if (i >= N_ * 32) return;
  int n = i >> 5, h = i & 31;
  int b = batch[n];
  unsafeAtomicAdd(&pooled[b * 32 + h], x[i]);
  if (h == 0) unsafeAtomicAdd(&counts[b], 1.0f);
}

__global__ void head_kernel(const float* __restrict__ pooled, const float* __restrict__ counts,
                            const float* __restrict__ hW1, const float* __restrict__ hb1,
                            const float* __restrict__ hW2, const float* __restrict__ hb2,
                            const float* __restrict__ hW3, const float* __restrict__ hb3,
                            const float* __restrict__ hW4, const float* __restrict__ hb4,
                            float* __restrict__ out) {
  int b = blockIdx.x * blockDim.x + threadIdx.x;
  if (b >= 64) return;
  float inv = 1.0f / fmaxf(counts[b], 1.0f);
  float a0[32];
  for (int h = 0; h < 32; h++) a0[h] = pooled[b * 32 + h] * inv;
  float a1[128];
  for (int j = 0; j < 128; j++) {
    float s = hb1[j];
    for (int k = 0; k < 32; k++) s = fmaf(a0[k], hW1[k * 128 + j], s);
    a1[j] = selu_f(s);
  }
  float a2[64];
  for (int j = 0; j < 64; j++) {
    float s = hb2[j];
    for (int k = 0; k < 128; k++) s = fmaf(a1[k], hW2[k * 64 + j], s);
    a2[j] = selu_f(s);
  }
  float a3[32];
  for (int j = 0; j < 32; j++) {
    float s = hb3[j];
    for (int k = 0; k < 64; k++) s = fmaf(a2[k], hW3[k * 32 + j], s);
    a3[j] = selu_f(s);
  }
  float o[21];
  for (int j = 0; j < 21; j++) {
    float s = hb4[j];
    for (int k = 0; k < 32; k++) s = fmaf(a3[k], hW4[k * 21 + j], s);
    o[j] = s;
  }
  const int idxt[36] = {0,1,2,3,4,5, 1,6,7,8,9,10, 2,7,11,12,13,14,
                        3,8,12,15,16,17, 4,9,13,16,18,19, 5,10,14,17,19,20};
  for (int t = 0; t < 36; t++) out[b * 36 + t] = o[idxt[t]];
}

// ---------------- fused edge kernel (WMMA, 32-edge tiles) ----------------
// One wave owns 32 edges (two M=16 fragments). Each W3T B-fragment from LDS
// feeds TWO WMMAs; biases ride in the WMMA C operand; the x_src tile is stored
// transposed so einsum weights load as broadcast ds_load_b128. The 32x1024
// per-edge weight matrix "w" never touches memory.
__global__ __launch_bounds__(EDGE_BLOCK_THREADS) void edge_kernel(
    const float* __restrict__ ef, const float* __restrict__ x, float* __restrict__ agg,
    const _Float16* __restrict__ w2t, const _Float16* __restrict__ w3t,
    const float* __restrict__ W1, const float* __restrict__ b1,
    const float* __restrict__ b2, const float* __restrict__ b3,
    const int* __restrict__ eidx, int E_, int ntiles) {
  extern __shared__ char smem[];
  _Float16* w3s = (_Float16*)smem;                       // 32768 f16
  _Float16* w2s = (_Float16*)(smem + 65536);             // 1024 f16
  float*    b3s = (float*)(smem + 65536 + 2048);         // 1024 f32
  char* wbase = smem + 65536 + 2048 + 4096;

  int wave = threadIdx.x >> 5;
  int lane = threadIdx.x & 31;
  char* wsc = wbase + wave * WAVE_SCRATCH;
  float*    xsT = (float*)wsc;                 // transposed x tile: xsT[h*32+m]
  _Float16* h1s = (_Float16*)(wsc + 4096);     // 32x32 f16
  _Float16* h2s = (_Float16*)(wsc + 6144);     // 32x32 f16
  float*    efs = (float*)(wsc + 8192);        // 32x5 f32
  int*      ids = (int*)(wsc + 8832);          // dst[32]

  // cooperative stage of layer weights into LDS
  {
    const uint4* g3 = (const uint4*)w3t;
    uint4* l3 = (uint4*)w3s;
    for (int i = threadIdx.x; i < 4096; i += EDGE_BLOCK_THREADS) l3[i] = g3[i];
    const uint4* g2 = (const uint4*)w2t;
    uint4* l2 = (uint4*)w2s;
    for (int i = threadIdx.x; i < 128; i += EDGE_BLOCK_THREADS) l2[i] = g2[i];
    for (int i = threadIdx.x; i < 1024; i += EDGE_BLOCK_THREADS) b3s[i] = b3[i];
  }
  __syncthreads();

  // hoisted per-lane constants (loop-invariant weights / biases / fragments)
  float w1c[5];
#pragma unroll
  for (int i = 0; i < 5; i++) w1c[i] = W1[i * 32 + lane];
  float b1j = b1[lane];
  const int nn = lane & 15;      // fragment column (N) / row id
  const int hi = lane >> 4;      // which half of the wave
  const int kbA = hi * 8;        // A-frag K base (documented split layout)
  const int kbB = hi * 8;        // B-frag K base (split layout, matches A)
  v8f cb2_0 = bias_frag(b2[nn]);          // b2 as WMMA C operand (cols 0..15)
  v8f cb2_1 = bias_frag(b2[16 + nn]);     // b2 as WMMA C operand (cols 16..31)
  v16h bW2_0 = load_b_frag(w2s + (0 * 16 + nn) * 32 + kbB);  // W2 B-frags stay
  v16h bW2_1 = load_b_frag(w2s + (1 * 16 + nn) * 32 + kbB);  // in registers

  int gw = blockIdx.x * WAVES_PER_BLOCK + wave;
  int nw = gridDim.x * WAVES_PER_BLOCK;
  for (int t = gw; t < ntiles; t += nw) {
    int eb = t * 32;
    // prefetch next tile's edge metadata (global_prefetch_b8)
    int tn = t + nw;
    if (tn < ntiles) {
      int en = tn * 32 + lane;
      if (en < E_) {
        __builtin_prefetch(&ef[(size_t)en * 5], 0, 1);
        __builtin_prefetch(&eidx[en], 0, 1);
        __builtin_prefetch(&eidx[E_ + en], 0, 1);
      }
    }
    // --- each lane owns one edge of the tile ---
    int e = eb + lane;
    int ec = e < E_ ? e : E_ - 1;
    int srcv = eidx[ec];
    ids[lane] = eidx[E_ + ec];                 // dst -> LDS (cross-lane scatter)
    {
      size_t o = (size_t)ec * 5;
#pragma unroll
      for (int i = 0; i < 5; i++) efs[lane * 5 + i] = ef[o + i];
    }
    // --- gather own x_src row, store transposed: xsT[h*32 + lane] ---
    {
      const float4* xr = (const float4*)(x + (size_t)srcv * 32);
      float rv[32];
#pragma unroll
      for (int q = 0; q < 8; q++) {
        float4 f = xr[q];
        rv[q * 4 + 0] = f.x; rv[q * 4 + 1] = f.y; rv[q * 4 + 2] = f.z; rv[q * 4 + 3] = f.w;
      }
#pragma unroll
      for (int h = 0; h < 32; h++) xsT[h * 32 + lane] = rv[h];
    }
    wave_lds_fence();
    // --- h1 = relu(ef @ W1 + b1), column-parallel (lane = column j) ---
#pragma unroll
    for (int m = 0; m < 32; m++) {
      float a = b1j;
#pragma unroll
      for (int i = 0; i < 5; i++) a = fmaf(efs[m * 5 + i], w1c[i], a);
      h1s[m * 32 + lane] = (_Float16)fmaxf(a, 0.f);
    }
    wave_lds_fence();
    // --- h2 = relu(h1 @ W2 + b2): 4 WMMAs, bias in C operand ---
    v16h aL = load_a_frag(h1s + nn * 32, kbA);          // edges 0..15
    v16h aH = load_a_frag(h1s + (16 + nn) * 32, kbA);   // edges 16..31
    v8f cL0 = __builtin_amdgcn_wmma_f32_16x16x32_f16(false, aL, false, bW2_0, (short)0, cb2_0, false, false);
    v8f cL1 = __builtin_amdgcn_wmma_f32_16x16x32_f16(false, aL, false, bW2_1, (short)0, cb2_1, false, false);
    v8f cH0 = __builtin_amdgcn_wmma_f32_16x16x32_f16(false, aH, false, bW2_0, (short)0, cb2_0, false, false);
    v8f cH1 = __builtin_amdgcn_wmma_f32_16x16x32_f16(false, aH, false, bW2_1, (short)0, cb2_1, false, false);
#pragma unroll
    for (int v = 0; v < 8; v++) {
      int m = v + 8 * hi;
      h2s[m * 32 + nn] = (_Float16)fmaxf(cL0[v], 0.f);
      h2s[m * 32 + 16 + nn] = (_Float16)fmaxf(cL1[v], 0.f);
      h2s[(16 + m) * 32 + nn] = (_Float16)fmaxf(cH0[v], 0.f);
      h2s[(16 + m) * 32 + 16 + nn] = (_Float16)fmaxf(cH1[v], 0.f);
    }
    wave_lds_fence();
    // --- w = relu(h2 @ W3 + b3) fused with msg = x_src . reshape(w,32,32) ---
    v16h a2L = load_a_frag(h2s + nn * 32, kbA);
    v16h a2H = load_a_frag(h2s + (16 + nn) * 32, kbA);
    v8f msgL0 = {}, msgL1 = {}, msgH0 = {}, msgH1 = {};
    union { float4 f4[2]; float f[8]; } xvL, xvH;
#pragma unroll 2
    for (int cb = 0; cb < 64; cb++) {
      int h = cb >> 1;                 // 16-col block maps to one matrix row h
      if ((cb & 1) == 0) {
        // broadcast loads: every lane in a half-wave reads the same 32B
        const float4* xp = (const float4*)(xsT + h * 32 + 8 * hi);
        xvL.f4[0] = xp[0]; xvL.f4[1] = xp[1];
        const float4* xq = (const float4*)(xsT + h * 32 + 16 + 8 * hi);
        xvH.f4[0] = xq[0]; xvH.f4[1] = xq[1];
      }
      v16h bF = load_b_frag(w3s + (cb * 16 + nn) * 32 + kbB);
      v8f cbias = bias_frag(b3s[cb * 16 + nn]);   // bias rides in C operand
      v8f cL = __builtin_amdgcn_wmma_f32_16x16x32_f16(false, a2L, false, bF, (short)0, cbias, false, false);
      v8f cH = __builtin_amdgcn_wmma_f32_16x16x32_f16(false, a2H, false, bF, (short)0, cbias, false, false);
      if (cb & 1) {
#pragma unroll
        for (int v = 0; v < 8; v++) {
          msgL1[v] = fmaf(xvL.f[v], fmaxf(cL[v], 0.f), msgL1[v]);
          msgH1[v] = fmaf(xvH.f[v], fmaxf(cH[v], 0.f), msgH1[v]);
        }
      } else {
#pragma unroll
        for (int v = 0; v < 8; v++) {
          msgL0[v] = fmaf(xvL.f[v], fmaxf(cL[v], 0.f), msgL0[v]);
          msgH0[v] = fmaf(xvH.f[v], fmaxf(cH[v], 0.f), msgH0[v]);
        }
      }
    }
    // --- scatter msg into agg[dst] with native f32 atomics ---
#pragma unroll
    for (int v = 0; v < 8; v++) {
      int m0 = v + 8 * hi;
      int e0 = eb + m0;
      if (e0 < E_) {
        int d = ids[m0];
        unsafeAtomicAdd(&agg[(size_t)d * 32 + nn], msgL0[v]);
        unsafeAtomicAdd(&agg[(size_t)d * 32 + 16 + nn], msgL1[v]);
      }
      int m1 = 16 + v + 8 * hi;
      int e1 = eb + m1;
      if (e1 < E_) {
        int d = ids[m1];
        unsafeAtomicAdd(&agg[(size_t)d * 32 + nn], msgH0[v]);
        unsafeAtomicAdd(&agg[(size_t)d * 32 + 16 + nn], msgH1[v]);
      }
    }
    wave_lds_fence();
  }
}

// ---------------- host side ----------------

static inline int cdiv(int a, int b) { return (a + b - 1) / b; }

extern "C" void kernel_launch(void* const* d_in, const int* in_sizes, int n_in,
                              void* d_out, int out_size, void* d_ws, size_t ws_size,
                              hipStream_t stream) {
  const float* node_attrs = (const float*)d_in[0];
  const float* positions  = (const float*)d_in[1];
  const float* shifts     = (const float*)d_in[2];
  const float* edge_attr  = (const float*)d_in[3];
  const int*   edge_index = (const int*)d_in[4];
  const int*   batch_ids  = (const int*)d_in[5];
  const float* embed_W    = (const float*)d_in[6];
  const float* embed_b    = (const float*)d_in[7];
  const float* e_W1 = (const float*)d_in[8];
  const float* e_b1 = (const float*)d_in[9];
  const float* e_W2 = (const float*)d_in[10];
  const float* e_b2 = (const float*)d_in[11];
  const float* e_W3 = (const float*)d_in[12];
  const float* e_b3 = (const float*)d_in[13];
  const float* conv_b = (const float*)d_in[14];
  const float* h_W1 = (const float*)d_in[15];
  const float* h_b1 = (const float*)d_in[16];
  const float* h_W2 = (const float*)d_in[17];
  const float* h_b2 = (const float*)d_in[18];
  const float* h_W3 = (const float*)d_in[19];
  const float* h_b3 = (const float*)d_in[20];
  const float* h_W4 = (const float*)d_in[21];
  const float* h_b4 = (const float*)d_in[22];
  float* out = (float*)d_out;

  const int N_ = in_sizes[0];       // node_attrs is (N,1)
  const int E_ = in_sizes[3];       // edge_attr is (E,1)

  // workspace carve-up
  char* p = (char*)d_ws;
  float* x = (float*)p;        p += (size_t)N_ * 32 * 4;
  float* agg = (float*)p;      p += (size_t)N_ * 32 * 4;
  float* ef = (float*)p;       p += (size_t)E_ * 5 * 4;
  float* pooled = (float*)p;   p += 64 * 32 * 4;
  float* counts = (float*)p;   p += 256;
  _Float16* w2t = (_Float16*)p; p += 3 * 1024 * 2;
  _Float16* w3t = (_Float16*)p; p += (size_t)3 * 32768 * 2;

  const int ntiles = cdiv(E_, 32);

  prep_weights<<<cdiv(3 * 1024 + 3 * 32768, 256), 256, 0, stream>>>(e_W2, e_W3, w2t, w3t);
  init_x_kernel<<<cdiv(N_ * 32, 256), 256, 0, stream>>>(node_attrs, embed_W, embed_b, x, N_);
  ef_kernel<<<cdiv(E_, 256), 256, 0, stream>>>(positions, shifts, edge_attr, edge_index, ef, E_);

  for (int layer = 0; layer < 3; layer++) {
    init_agg_kernel<<<cdiv(N_ * 32, 256), 256, 0, stream>>>(agg, conv_b, layer, N_);
    edge_kernel<<<512, EDGE_BLOCK_THREADS, SMEM_EDGE, stream>>>(
        ef, x, agg,
        w2t + (size_t)layer * 1024, w3t + (size_t)layer * 32768,
        e_W1 + (size_t)layer * 5 * 32, e_b1 + (size_t)layer * 32,
        e_b2 + (size_t)layer * 32, e_b3 + (size_t)layer * 1024,
        edge_index, E_, ntiles);
    update_x_kernel<<<cdiv(N_ * 32, 256), 256, 0, stream>>>(x, agg, N_);
  }

  init_pool_kernel<<<cdiv(64 * 32, 256), 256, 0, stream>>>(pooled, counts);
  pool_kernel<<<cdiv(N_ * 32, 256), 256, 0, stream>>>(x, batch_ids, pooled, counts, N_);
  head_kernel<<<1, 64, 0, stream>>>(pooled, counts, h_W1, h_b1, h_W2, h_b2,
                                    h_W3, h_b3, h_W4, h_b4, out);
}
